// PanguAttentionRmPad_53472342835461
// MI455X (gfx1250) — compile-verified, matrix operand order checked by reading, not run on hardware
//
#include <hip/hip_runtime.h>
#include <hip/hip_bf16.h>

// ---------------------------------------------------------------------------
// PanguAttentionRmPad for MI455X (gfx1250): full f16-WMMA pipeline.
// T=3072, HID=1024, H=16, D=64.
// ---------------------------------------------------------------------------

#define T_TOK 3072
#define HID   1024
#define NH    16
#define HD    64

typedef __attribute__((ext_vector_type(8)))  _Float16 v8h;
typedef __attribute__((ext_vector_type(16))) _Float16 v16h;
typedef __attribute__((ext_vector_type(8)))  float    v8f;

// Load one f16 WMMA fragment (A or B pattern; both use the same per-lane
// pattern when the source is stored [row][K] with this lane's row folded into
// `p` already): elements 0..7 = K chunk at p, elements 8..15 = K chunk at p+16.
__device__ __forceinline__ v16h load_frag16(const _Float16* p) {
    v8h lo = *(const v8h*)p;
    v8h hi = *(const v8h*)(p + 16);
    v16h f;
#pragma unroll
    for (int i = 0; i < 8; ++i) { f[i] = lo[i]; f[i + 8] = hi[i]; }
    return f;
}

__device__ __forceinline__ v8f wmma_f16(v16h a, v16h b, v8f c) {
    // (neg_a, A, neg_b, B, c_mod, C, reuse_a, reuse_b)
    return __builtin_amdgcn_wmma_f32_16x16x32_f16(false, a, false, b,
                                                  (short)0, c, false, false);
}

// ---------------------------------------------------------------------------
// Conversions
// ---------------------------------------------------------------------------
__global__ void k_f32_to_f16(const float* __restrict__ in,
                             _Float16* __restrict__ out, int n) {
    int i = blockIdx.x * blockDim.x + threadIdx.x;
    if (i < n) out[i] = (_Float16)in[i];
}

// in: [K][N] row-major f32; out: [N][K] f16 (so B-fragments read contiguous K)
__global__ void k_transpose_f32_to_f16(const float* __restrict__ in,
                                       _Float16* __restrict__ out,
                                       int K, int N) {
    int idx = blockIdx.x * blockDim.x + threadIdx.x;
    if (idx >= K * N) return;
    int k = idx % K;
    int n = idx / K;
    out[(size_t)n * K + k] = (_Float16)in[(size_t)k * N + n];
}

// ---------------------------------------------------------------------------
// Tiled WMMA GEMM: C[M,N] = A[M,K] * Bt[N,K]^T ; A,Bt f16, accumulate f32.
// Block = 128 threads (4 waves); block tile = 16 rows x 256 cols,
// each wave owns a 16x64 strip (4 n-tiles), K-loop in chunks of 32.
// Unconditional prefetch of the next k-chunk: the one-chunk overrun past the
// panel end stays inside the workspace allocation (speculative prefetches are
// dropped harmlessly per ISA; no data/counters involved).
// ---------------------------------------------------------------------------
template <bool OUT_HALF>
__global__ __launch_bounds__(128) void k_gemm_f16(
    const _Float16* __restrict__ A, const _Float16* __restrict__ Bt,
    int N, int K, _Float16* __restrict__ Ch, float* __restrict__ Cf) {
    const int lane = threadIdx.x & 31;
    const int wave = threadIdx.x >> 5;
    const int nn   = lane & 15;
    const int half = lane >> 4;
    const int r0   = blockIdx.x * 16;
    const int n0   = blockIdx.y * 256 + wave * 64;

    v8f acc0 = {}, acc1 = {}, acc2 = {}, acc3 = {};
    const _Float16* Ap = A + (size_t)(r0 + nn) * K;
    const _Float16* B0 = Bt + (size_t)(n0 +  0 + nn) * K;
    const _Float16* B1 = Bt + (size_t)(n0 + 16 + nn) * K;
    const _Float16* B2 = Bt + (size_t)(n0 + 32 + nn) * K;
    const _Float16* B3 = Bt + (size_t)(n0 + 48 + nn) * K;

    for (int k = 0; k < K; k += 32) {
        const int ko = k + half * 8;
        __builtin_prefetch(Ap + ko + 32, 0, 1);
        __builtin_prefetch(B0 + ko + 32, 0, 1);
        __builtin_prefetch(B1 + ko + 32, 0, 1);
        __builtin_prefetch(B2 + ko + 32, 0, 1);
        __builtin_prefetch(B3 + ko + 32, 0, 1);
        v16h a = load_frag16(Ap + ko);
        v16h b;
        b = load_frag16(B0 + ko); acc0 = wmma_f16(a, b, acc0);
        b = load_frag16(B1 + ko); acc1 = wmma_f16(a, b, acc1);
        b = load_frag16(B2 + ko); acc2 = wmma_f16(a, b, acc2);
        b = load_frag16(B3 + ko); acc3 = wmma_f16(a, b, acc3);
    }

#pragma unroll
    for (int r = 0; r < 8; ++r) {
        int row = r0 + r + 8 * half;   // C/D layout: lanes 16-31 hold M=8+r
        if (OUT_HALF) {
            _Float16* p = Ch + (size_t)row * N + n0;
            p[ 0 + nn] = (_Float16)acc0[r];
            p[16 + nn] = (_Float16)acc1[r];
            p[32 + nn] = (_Float16)acc2[r];
            p[48 + nn] = (_Float16)acc3[r];
        } else {
            float* p = Cf + (size_t)row * N + n0;
            p[ 0 + nn] = acc0[r];
            p[16 + nn] = acc1[r];
            p[32 + nn] = acc2[r];
            p[48 + nn] = acc3[r];
        }
    }
}

// ---------------------------------------------------------------------------
// RoPE + repack. qkv: [T, 3*HID] f16.  One thread per (t, h, i), i in [0,32):
// rotates pair (d=i, d=i+32). Writes Qh/Kh as [h][t][d] (A / score-B layout)
// and V transposed as Vt [h][d][t] (B layout for P·V).
// ---------------------------------------------------------------------------
__global__ void k_rope_pack(const _Float16* __restrict__ qkv,
                            const int* __restrict__ pos_ids,
                            _Float16* __restrict__ Qh,
                            _Float16* __restrict__ Kh,
                            _Float16* __restrict__ Vt) {
    int idx = blockIdx.x * blockDim.x + threadIdx.x;  // T*NH*32 threads
    int i = idx & 31;
    int h = (idx >> 5) & (NH - 1);
    int t = idx >> 9;

    float pos  = (float)pos_ids[t];
    // inv_freq[i] = theta^(-2i/D) = exp(-(2i/D)*ln(theta))
    float invf = __expf(-((float)(2 * i) / (float)HD) * 9.210340371976184f);
    float fr   = pos * invf;
    float s, c;
    __sincosf(fr, &s, &c);

    const _Float16* q = qkv + (size_t)t * (3 * HID) + h * HD;
    const _Float16* k = q + HID;
    const _Float16* v = q + 2 * HID;
    float q0 = (float)q[i], q1 = (float)q[i + 32];
    float k0 = (float)k[i], k1 = (float)k[i + 32];

    _Float16* Qp = Qh + ((size_t)h * T_TOK + t) * HD;
    Qp[i]      = (_Float16)(q0 * c - q1 * s);
    Qp[i + 32] = (_Float16)(q1 * c + q0 * s);
    _Float16* Kp = Kh + ((size_t)h * T_TOK + t) * HD;
    Kp[i]      = (_Float16)(k0 * c - k1 * s);
    Kp[i + 32] = (_Float16)(k1 * c + k0 * s);

    Vt[((size_t)h * HD + i)      * T_TOK + t] = v[i];
    Vt[((size_t)h * HD + i + 32) * T_TOK + t] = v[i + 32];
}

// ---------------------------------------------------------------------------
// Flash attention: one wave per (head, 16 q-rows). Online softmax, scores
// never leave registers; P transposed through 1KB LDS to become an A-fragment.
// Unconditional prefetch of next K/V/mask tiles overlaps L2 fills with the
// ~60 VALU ops of softmax per iteration.
// ---------------------------------------------------------------------------
__global__ __launch_bounds__(32) void k_flash_attn(
    const _Float16* __restrict__ Qh, const _Float16* __restrict__ Kh,
    const _Float16* __restrict__ Vt, const float* __restrict__ mask,
    _Float16* __restrict__ attn) {
    __shared__ __align__(16) _Float16 pl[16 * 32];

    const int lane = threadIdx.x & 31;
    const int nn   = lane & 15;
    const int half = lane >> 4;
    const int h    = blockIdx.y;
    const int q0   = blockIdx.x * 16;

    const _Float16* Qp = Qh + ((size_t)h * T_TOK + q0) * HD;
    const _Float16* Kp = Kh + (size_t)h * T_TOK * HD;
    const _Float16* Vp = Vt + (size_t)h * HD * T_TOK;

    // Q A-fragments (rows q0..q0+15, d-chunks 0..31 and 32..63), loaded once.
    v16h aq0 = load_frag16(Qp + nn * HD +  0 + half * 8);
    v16h aq1 = load_frag16(Qp + nn * HD + 32 + half * 8);

    float mrow[8], lrow[8];
    v8f o0 = {}, o1 = {}, o2 = {}, o3 = {};
#pragma unroll
    for (int r = 0; r < 8; ++r) { mrow[r] = -1e30f; lrow[r] = 0.0f; }

    const float scale = 0.125f;  // 1/sqrt(64)

    for (int j = 0; j < T_TOK; j += 32) {
        // ----- prefetch next key/value/mask tiles -----
        __builtin_prefetch(Kp + (size_t)(j + 32 + nn) * HD + half * 8, 0, 1);
        __builtin_prefetch(Kp + (size_t)(j + 48 + nn) * HD + half * 8, 0, 1);
        __builtin_prefetch(Vp + (size_t)(nn)      * T_TOK + j + 32 + half * 8, 0, 1);
        __builtin_prefetch(Vp + (size_t)(32 + nn) * T_TOK + j + 32 + half * 8, 0, 1);
        __builtin_prefetch(mask + (size_t)(q0 + nn) * T_TOK + j + 32, 0, 1);

        // ----- scores for keys j..j+31 (two 16-key column tiles) -----
        v16h bk;
        v8f s0 = {}, s1 = {};
        bk = load_frag16(Kp + (size_t)(j      + nn) * HD +  0 + half * 8);
        s0 = wmma_f16(aq0, bk, s0);
        bk = load_frag16(Kp + (size_t)(j      + nn) * HD + 32 + half * 8);
        s0 = wmma_f16(aq1, bk, s0);
        bk = load_frag16(Kp + (size_t)(j + 16 + nn) * HD +  0 + half * 8);
        s1 = wmma_f16(aq0, bk, s1);
        bk = load_frag16(Kp + (size_t)(j + 16 + nn) * HD + 32 + half * 8);
        s1 = wmma_f16(aq1, bk, s1);

        // scale + additive mask (row = q0 + r + 8*half, col = j + {nn, 16+nn})
        float p0[8], p1[8], mnew[8], corr[8];
#pragma unroll
        for (int r = 0; r < 8; ++r) {
            int row = q0 + r + 8 * half;
            p0[r] = s0[r] * scale + mask[(size_t)row * T_TOK + j + nn];
            p1[r] = s1[r] * scale + mask[(size_t)row * T_TOK + j + 16 + nn];
        }
        // row max across the 16 lanes of this half (masks <16 stay in-half)
#pragma unroll
        for (int r = 0; r < 8; ++r) {
            float t = fmaxf(p0[r], p1[r]);
#pragma unroll
            for (int off = 1; off < 16; off <<= 1)
                t = fmaxf(t, __shfl_xor(t, off, 32));
            mnew[r] = fmaxf(mrow[r], t);
        }
#pragma unroll
        for (int r = 0; r < 8; ++r) {
            corr[r] = __expf(mrow[r] - mnew[r]);
            p0[r]   = __expf(p0[r] - mnew[r]);
            p1[r]   = __expf(p1[r] - mnew[r]);
            float rs = p0[r] + p1[r];
#pragma unroll
            for (int off = 1; off < 16; off <<= 1)
                rs += __shfl_xor(rs, off, 32);
            lrow[r] = lrow[r] * corr[r] + rs;
            mrow[r] = mnew[r];
        }
#pragma unroll
        for (int r = 0; r < 8; ++r) {
            o0[r] *= corr[r]; o1[r] *= corr[r];
            o2[r] *= corr[r]; o3[r] *= corr[r];
        }

        // ----- transpose P (C layout -> A fragment) through LDS -----
        __syncthreads();
#pragma unroll
        for (int r = 0; r < 8; ++r) {
            int row = r + 8 * half;
            pl[row * 32 + nn]      = (_Float16)p0[r];
            pl[row * 32 + 16 + nn] = (_Float16)p1[r];
        }
        __syncthreads();
        v16h ap = load_frag16(&pl[nn * 32 + half * 8]);

        // ----- P @ V : 4 n-tiles of d -----
        v16h bv;
        bv = load_frag16(Vp + (size_t)( 0 + nn) * T_TOK + j + half * 8);
        o0 = wmma_f16(ap, bv, o0);
        bv = load_frag16(Vp + (size_t)(16 + nn) * T_TOK + j + half * 8);
        o1 = wmma_f16(ap, bv, o1);
        bv = load_frag16(Vp + (size_t)(32 + nn) * T_TOK + j + half * 8);
        o2 = wmma_f16(ap, bv, o2);
        bv = load_frag16(Vp + (size_t)(48 + nn) * T_TOK + j + half * 8);
        o3 = wmma_f16(ap, bv, o3);
    }

    // normalize and store into attn [T, HID] f16
#pragma unroll
    for (int r = 0; r < 8; ++r) {
        float inv = 1.0f / lrow[r];
        int row = q0 + r + 8 * half;
        _Float16* op = attn + (size_t)row * HID + h * HD;
        op[ 0 + nn] = (_Float16)(o0[r] * inv);
        op[16 + nn] = (_Float16)(o1[r] * inv);
        op[32 + nn] = (_Float16)(o2[r] * inv);
        op[48 + nn] = (_Float16)(o3[r] * inv);
    }
}

// ---------------------------------------------------------------------------
// Launch
// ---------------------------------------------------------------------------
extern "C" void kernel_launch(void* const* d_in, const int* in_sizes, int n_in,
                              void* d_out, int out_size, void* d_ws, size_t ws_size,
                              hipStream_t stream) {
    const float* hidden = (const float*)d_in[0];  // [T, HID]
    const float* w_qkv  = (const float*)d_in[1];  // [HID, 3*HID]
    const float* w_o    = (const float*)d_in[2];  // [HID, HID]
    const float* mask   = (const float*)d_in[3];  // [T, T]
    const int*   pos    = (const int*)d_in[4];    // [T]
    float*       out    = (float*)d_out;          // [T, HID]

    char* ws = (char*)d_ws;
    _Float16* Xh    = (_Float16*)ws; ws += (size_t)T_TOK * HID * 2;       // 6 MB
    _Float16* Wqt   = (_Float16*)ws; ws += (size_t)(3 * HID) * HID * 2;   // 6 MB
    _Float16* Wot   = (_Float16*)ws; ws += (size_t)HID * HID * 2;         // 2 MB
    _Float16* QKV   = (_Float16*)ws; ws += (size_t)T_TOK * 3 * HID * 2;   // 18 MB
    _Float16* Qh    = (_Float16*)ws; ws += (size_t)NH * T_TOK * HD * 2;   // 6 MB
    _Float16* Kh    = (_Float16*)ws; ws += (size_t)NH * T_TOK * HD * 2;   // 6 MB
    _Float16* Vt    = (_Float16*)ws; ws += (size_t)NH * T_TOK * HD * 2;   // 6 MB
    _Float16* attnh = (_Float16*)ws; ws += (size_t)T_TOK * HID * 2;       // 6 MB

    // 1) precision conversion / weight transposition
    {
        int n = T_TOK * HID;
        k_f32_to_f16<<<(n + 255) / 256, 256, 0, stream>>>(hidden, Xh, n);
    }
    {
        int n = HID * 3 * HID;
        k_transpose_f32_to_f16<<<(n + 255) / 256, 256, 0, stream>>>(
            w_qkv, Wqt, HID, 3 * HID);
    }
    {
        int n = HID * HID;
        k_transpose_f32_to_f16<<<(n + 255) / 256, 256, 0, stream>>>(
            w_o, Wot, HID, HID);
    }

    // 2) QKV projection: [T,HID] x [HID,3*HID] -> f16 [T, 3*HID]
    k_gemm_f16<true><<<dim3(T_TOK / 16, (3 * HID) / 256), 128, 0, stream>>>(
        Xh, Wqt, 3 * HID, HID, QKV, nullptr);

    // 3) RoPE + repack into per-head layouts
    {
        int n = T_TOK * NH * 32;
        k_rope_pack<<<n / 256, 256, 0, stream>>>(QKV, pos, Qh, Kh, Vt);
    }

    // 4) flash attention (per-head, 16 q-rows per wave)
    k_flash_attn<<<dim3(T_TOK / 16, NH), 32, 0, stream>>>(Qh, Kh, Vt, mask, attnh);

    // 5) output projection: [T,HID] x [HID,HID] -> f32 d_out
    k_gemm_f16<false><<<dim3(T_TOK / 16, HID / 256), 128, 0, stream>>>(
        attnh, Wot, HID, HID, nullptr, out);
}
